// GwACIter_28123445854583
// MI455X (gfx1250) — matrix operand order repaired and checked by reading, not compiled, not run on hardware
//
#include <hip/hip_runtime.h>
#include <hip/hip_bf16.h>
#include <math.h>

// ---------------- problem constants (from reference) ----------------
#define NNODE  256
#define INFEAT 64
#define HDIM   256
#define MDIM   128
#define OUTD   32
#define PNUM   2
#define DEG    8
#define SSTART 16
#define CAPMSG (10 * NNODE)            // 2560
#define QLEN   (SSTART + CAPMSG * DEG) // 20496
#define KDIM   (HDIM + MDIM)           // 384
#define WPAD   392                     // padded LDS column stride (halfs)

// ---------------- workspace layout (floats) ----------------
#define FEAT_OFF   0u            // 256*256
#define FINAL_OFF  65536u        // 256*256
#define LOGIT_OFF  131072u       // 2*256*32 = 16384
#define QMSG_OFF   147456u       // QLEN*128 = 2623488 (16B-aligned rows)
#define QNODE_OFF  2770944u      // QLEN ints

// ---------------- LDS layout (bytes) for persistent kernel ----------------
// dynamic LDS starts at offset 0 (no static __shared__ in persist_kernel),
// so these byte offsets are also the LDS addresses used by the async engine.
#define L_WST   0u                           // 256 * 392 * 2 = 200704
#define L_WC    200704u                      // 384 * 4
#define L_BST   202240u                      // 256 * 4
#define L_BMG   203264u                      // 128 * 4
#define L_FM    203776u                      // 384 * 4  (feat row + msg)
#define L_NSMSG 205312u                      // 384 * 4
#define L_NSRAW 206848u                      // 256 * 4
#define L_NMRAW 207872u                      // 128 * 4  (final nm incl. bias)
#define L_REM   208384u                      // 256 * 4
#define L_Z     209408u                      // 16
#define SMEM_BYTES 209424u                   // well under 320KB WGP LDS

typedef __attribute__((ext_vector_type(16))) __bf16 v16bf;
typedef __attribute__((ext_vector_type(8)))  __bf16 v8bf;
typedef __attribute__((ext_vector_type(8)))  float  v8f;

__device__ __forceinline__ v8f wmma_bf16(v16bf a, v16bf b, v8f c) {
  // (neg_a, A, neg_b, B, c_mod, C, reuse_a, reuse_b)
  return __builtin_amdgcn_wmma_f32_16x16x32_bf16(false, a, false, b, (short)0, c,
                                                 false, false);
}

// A-fragment (16x32 bf16) from row-major f32 matrix; lane L: M=L%16,
// elem i -> K = kt*32 + (L/16)*16 + i
__device__ __forceinline__ v16bf load_a_f32(const float* base, int ld, int row0,
                                            int kt, int lane) {
  int m = lane & 15, hi = lane >> 4;
  const float* p = base + (size_t)(row0 + m) * ld + kt * 32 + hi * 16;
  v16bf a;
#pragma unroll
  for (int i = 0; i < 8; ++i) {
    a[i]     = (__bf16)p[i];
    a[i + 8] = (__bf16)p[8 + i];
  }
  return a;
}

// A-fragment with only row 0 valid (GEMV): vector is f32 in LDS
__device__ __forceinline__ v16bf load_a_vec0(const float* vec, int kt, int lane) {
  v16bf a;
#pragma unroll
  for (int i = 0; i < 16; ++i) a[i] = (__bf16)0.0f;
  if ((lane & 15) == 0) {
    int hi = lane >> 4;
    const float* p = vec + kt * 32 + hi * 16;
#pragma unroll
    for (int i = 0; i < 8; ++i) {
      a[i]     = (__bf16)p[i];
      a[i + 8] = (__bf16)p[8 + i];
    }
  }
  return a;
}

// B-fragment (32x16 bf16) from row-major f32 matrix; lane L: N=L%16,
// elem i -> K = kt*32 + (L/16)*16 + i
__device__ __forceinline__ v16bf load_b_f32(const float* base, int ld, int col0,
                                            int kt, int lane) {
  int n = lane & 15, hi = lane >> 4;
  const float* p = base + (size_t)(kt * 32 + hi * 16) * ld + col0 + n;
  v16bf b;
#pragma unroll
  for (int i = 0; i < 16; ++i) b[i] = (__bf16)p[(size_t)i * ld];
  return b;
}

// B-fragment from LDS, weights stored column-major bf16 with padded stride
__device__ __forceinline__ v16bf load_b_lds(const __bf16* W, int stride, int col0,
                                            int kt, int lane) {
  int n = lane & 15, hi = lane >> 4;
  const __bf16* p = W + (size_t)(col0 + n) * stride + kt * 32 + hi * 16;
  const v8bf* p8 = (const v8bf*)p;           // 16B aligned by construction
  v8bf lo = p8[0], hv = p8[1];
  return __builtin_shufflevector(lo, hv, 0, 1, 2, 3, 4, 5, 6, 7, 8, 9, 10, 11,
                                 12, 13, 14, 15);
}

// ================= encoder: feat = x @ W_enc + b_enc (WMMA GEMM) =================
__global__ void enc_kernel(const float* __restrict__ x,
                           const float* __restrict__ W_enc,
                           const float* __restrict__ b_enc,
                           float* __restrict__ feat) {
  int t = threadIdx.x, wid = t >> 5, lane = t & 31;
  int gid = blockIdx.x * 8 + wid;   // 256 tiles
  int mt = gid >> 4, nt = gid & 15;
  int n = lane & 15, hi = lane >> 4;
  v8f acc = {};
#pragma unroll
  for (int kt = 0; kt < INFEAT / 32; ++kt) {
    v16bf a = load_a_f32(x, INFEAT, mt * 16, kt, lane);
    v16bf b = load_b_f32(W_enc, HDIM, nt * 16, kt, lane);
    acc = wmma_bf16(a, b, acc);
  }
  float bias = b_enc[nt * 16 + n];
#pragma unroll
  for (int j = 0; j < 8; ++j) {
    int row = mt * 16 + j + 8 * hi;   // D layout: VGPR j -> M=j (lanes<16) / j+8
    feat[(size_t)row * HDIM + nt * 16 + n] = acc[j] + bias;
  }
}

// ================= persistent sequential ACT loop =================
__global__ void __launch_bounds__(256, 1)
persist_kernel(const float* __restrict__ first_message,
               const int*   __restrict__ nbrs,
               const float* __restrict__ W_state, const float* __restrict__ b_state,
               const float* __restrict__ W_msg,   const float* __restrict__ b_msg,
               const float* __restrict__ W_conf,  const float* __restrict__ b_conf,
               float* __restrict__ featp, float* __restrict__ finalp,
               float* __restrict__ qmsgs, int* __restrict__ qnodes) {
  extern __shared__ char smem[];
  __bf16* Wst   = (__bf16*)(smem + L_WST);
  float*  wc    = (float*)(smem + L_WC);
  float*  bst   = (float*)(smem + L_BST);
  float*  bmg   = (float*)(smem + L_BMG);
  float*  fm    = (float*)(smem + L_FM);
  float*  nsmsg = (float*)(smem + L_NSMSG);
  float*  nsraw = (float*)(smem + L_NSRAW);
  float*  nmraw = (float*)(smem + L_NMRAW);
  float*  rem   = (float*)(smem + L_REM);
  float*  zslot = (float*)(smem + L_Z);

  const int t = threadIdx.x, wid = t >> 5, lane = t & 31;

  // ---- stage W_state into LDS as bf16 (column-major, padded stride) ----
  for (int e = t; e < KDIM * HDIM; e += 256) {
    int k = e >> 8, c = e & 255;
    Wst[(size_t)c * WPAD + k] = (__bf16)W_state[e];
  }
  for (int e = t; e < KDIM; e += 256) wc[e] = W_conf[e];
  bst[t] = b_state[t];
  rem[t] = 1.0f;
  if (t < MDIM) bmg[t] = b_msg[t];
  // final = 0
  for (int e = t; e < NNODE * HDIM; e += 256) finalp[e] = 0.0f;
  // queue seed: start nodes 0..15 with their first_message
  if (t < SSTART) qnodes[t] = t;
  for (int e = t; e < SSTART * MDIM; e += 256) qmsgs[e] = first_message[e];

  // ---- W_msg B-fragments live in registers for the whole loop ----
  // wave w owns output tile nt = w (16 cols); 12 K-frags = 96 VGPRs of bf16
  v16bf wm[KDIM / 32];
#pragma unroll
  for (int kt = 0; kt < KDIM / 32; ++kt)
    wm[kt] = load_b_f32(W_msg, MDIM, wid * 16, kt, lane);

  __threadfence_block();
  __syncthreads();

  const float bconf0 = b_conf[0];
  int head = 0, tail = SSTART, cnt = 0;

  for (int iter = 0; iter < QLEN; ++iter) {
    if (!(head < tail && cnt < CAPMSG)) break;
    int node = qnodes[head];
    float r = rem[node];
    bool active = (r >= 1e-7f);            // uniform across the workgroup

    if (active) {
      // ---- async-stage fm = concat(feat[node], msg) straight into LDS ----
      if (t == 0) zslot[0] = 0.0f;
      if (t < 96) {
        unsigned ldsoff;
        const char* g;
        if (t < 64) {  // feat row: 64 lanes x 16B = 1KB
          ldsoff = L_FM + (unsigned)t * 16u;
          g = (const char*)(featp + (size_t)node * HDIM) + t * 16;
        } else {       // message row: 32 lanes x 16B = 512B
          int u = t - 64;
          ldsoff = L_FM + HDIM * 4u + (unsigned)u * 16u;
          g = (const char*)(qmsgs + (size_t)head * MDIM) + u * 16;
        }
        asm volatile("global_load_async_to_lds_b128 %0, %1, off"
                     :: "v"(ldsoff), "v"(g) : "memory");
      }
      asm volatile("s_wait_asynccnt 0x0" ::: "memory");
      __syncthreads();

      // ---- confidence dot: shfl reduce + one LDS atomic per wave ----
      float pz = fm[t] * wc[t];
      if (t < MDIM) pz += fm[HDIM + t] * wc[HDIM + t];
#pragma unroll
      for (int o = 16; o > 0; o >>= 1) pz += __shfl_xor(pz, o, 32);
      if (lane == 0) atomicAdd(&zslot[0], pz);

      // ---- ns = fm @ W_state + b_state (bias folded into C row 0) ----
#pragma unroll
      for (int nti = 0; nti < 2; ++nti) {
        int nt = wid + nti * 8;
        v8f acc = {};
        if (lane < 16) acc[0] = bst[nt * 16 + lane];
#pragma unroll
        for (int kt = 0; kt < KDIM / 32; ++kt) {
          v16bf a = load_a_vec0(fm, kt, lane);
          v16bf b = load_b_lds(Wst, WPAD, nt * 16, kt, lane);
          acc = wmma_bf16(a, b, acc);
        }
        if (lane < 16) nsraw[nt * 16 + lane] = acc[0];  // D row 0 = lanes 0-15
      }
      __syncthreads();

      float z  = zslot[0] + bconf0;
      float sc = 1.0f / (1.0f + expf(-z));
      float nsv = nsraw[t];
      nsv = nsv > 0.0f ? nsv : 0.0f;                    // relu (bias already in)
      nsmsg[t] = nsv;                                   // concat(ns, msg)
      if (t < MDIM) nsmsg[HDIM + t] = fm[HDIM + t];
      __syncthreads();

      // ---- nm = concat(ns,msg) @ W_msg + b_msg (register-resident B) ----
      {
        v8f acc = {};
        if (lane < 16) acc[0] = bmg[wid * 16 + lane];
#pragma unroll
        for (int kt = 0; kt < KDIM / 32; ++kt) {
          v16bf a = load_a_vec0(nsmsg, kt, lane);
          acc = wmma_bf16(a, wm[kt], acc);
        }
        if (lane < 16) nmraw[wid * 16 + lane] = acc[0]; // final value incl bias
      }
      __syncthreads();

      // ---- prefetch next pop (overlaps with the commit stores) ----
      if (head + 1 < tail) {
        int nn = qnodes[head + 1];
        __builtin_prefetch(featp + (size_t)nn * HDIM, 0, 0);
        __builtin_prefetch(qmsgs + (size_t)(head + 1) * MDIM, 0, 0);
      }

      // ---- commit node updates ----
      featp[(size_t)node * HDIM + t] = nsv;
      finalp[(size_t)node * HDIM + t] += r * sc * nsv;
      if (t == 0) rem[node] = r * (1.0f - sc);
      if (t < DEG) qnodes[tail + t] = nbrs[node * DEG + t];

      // ---- enqueue: broadcast nm (512B in LDS) to 8 queue rows, async ----
      {
        int d = t >> 5, j16 = t & 31;                    // 8 rows x 32 x 16B
        unsigned ldsoff = L_NMRAW + (unsigned)j16 * 16u;
        const char* g =
            (const char*)(qmsgs + (size_t)(tail + d) * MDIM) + j16 * 16;
        asm volatile("global_store_async_from_lds_b128 %0, %1, off"
                     :: "v"(g), "v"(ldsoff) : "memory");
      }
      asm volatile("s_wait_asynccnt 0x0" ::: "memory");
      __threadfence_block();
      __syncthreads();
      tail += DEG;
      cnt += 1;
    }
    head += 1;   // pop even on skipped entries
  }
}

// ================= decoder: logits[p] = final @ W_dec[p] + b_dec[p] =================
__global__ void dec_kernel(const float* __restrict__ finalp,
                           const float* __restrict__ W_dec,
                           const float* __restrict__ b_dec,
                           float* __restrict__ logits) {
  int t = threadIdx.x, wid = t >> 5, lane = t & 31;
  int gid = blockIdx.x * 8 + wid;        // 64 tiles: P(2) x Mtiles(16) x Ntiles(2)
  int p = gid >> 5, rr = gid & 31, mt = rr >> 1, nt2 = rr & 1;
  int n = lane & 15, hi = lane >> 4;
  const float* Wd = W_dec + (size_t)p * HDIM * OUTD;
  v8f acc = {};
#pragma unroll
  for (int kt = 0; kt < HDIM / 32; ++kt) {
    v16bf a = load_a_f32(finalp, HDIM, mt * 16, kt, lane);
    v16bf b = load_b_f32(Wd, OUTD, nt2 * 16, kt, lane);
    acc = wmma_bf16(a, b, acc);
  }
  float bias = b_dec[p * OUTD + nt2 * 16 + n];
#pragma unroll
  for (int j = 0; j < 8; ++j) {
    int row = mt * 16 + j + 8 * hi;
    logits[(size_t)p * NNODE * OUTD + (size_t)row * OUTD + nt2 * 16 + n] =
        acc[j] + bias;
  }
}

// ================= log_softmax over OUT=32 (one wave32 per row) =================
__global__ void lsm_kernel(const float* __restrict__ logits, float* __restrict__ out) {
  int t = threadIdx.x, wid = t >> 5, lane = t & 31;
  int row = blockIdx.x * 8 + wid;        // 512 rows = P*N
  float v = logits[(size_t)row * OUTD + lane];
  float m = v;
#pragma unroll
  for (int o = 16; o > 0; o >>= 1) m = fmaxf(m, __shfl_xor(m, o, 32));
  float e = expf(v - m), s = e;
#pragma unroll
  for (int o = 16; o > 0; o >>= 1) s += __shfl_xor(s, o, 32);
  out[(size_t)row * OUTD + lane] = (v - m) - logf(s);
}

// ================= host launcher =================
extern "C" void kernel_launch(void* const* d_in, const int* in_sizes, int n_in,
                              void* d_out, int out_size, void* d_ws, size_t ws_size,
                              hipStream_t stream) {
  (void)in_sizes; (void)n_in; (void)out_size; (void)ws_size;
  const float* x        = (const float*)d_in[0];
  const float* firstmsg = (const float*)d_in[1];
  const int*   nbrs     = (const int*)  d_in[2];
  const float* W_enc    = (const float*)d_in[3];
  const float* b_enc    = (const float*)d_in[4];
  const float* W_state  = (const float*)d_in[5];
  const float* b_state  = (const float*)d_in[6];
  const float* W_msg    = (const float*)d_in[7];
  const float* b_msg    = (const float*)d_in[8];
  const float* W_conf   = (const float*)d_in[9];
  const float* b_conf   = (const float*)d_in[10];
  const float* W_dec    = (const float*)d_in[11];
  const float* b_dec    = (const float*)d_in[12];

  float* wsf    = (float*)d_ws;
  float* feat   = wsf + FEAT_OFF;
  float* finalp = wsf + FINAL_OFF;
  float* logits = wsf + LOGIT_OFF;
  float* qmsgs  = wsf + QMSG_OFF;
  int*   qnodes = (int*)(wsf + QNODE_OFF);

  // allow ~205KB dynamic LDS for the persistent kernel (WGP has 320KB)
  (void)hipFuncSetAttribute((const void*)persist_kernel,
                            hipFuncAttributeMaxDynamicSharedMemorySize,
                            (int)SMEM_BYTES);

  enc_kernel<<<32, 256, 0, stream>>>(x, W_enc, b_enc, feat);
  persist_kernel<<<1, 256, SMEM_BYTES, stream>>>(firstmsg, nbrs, W_state, b_state,
                                                 W_msg, b_msg, W_conf, b_conf,
                                                 feat, finalp, qmsgs, qnodes);
  dec_kernel<<<8, 256, 0, stream>>>(finalp, W_dec, b_dec, logits);
  lsm_kernel<<<64, 256, 0, stream>>>(logits, (float*)d_out);
}
